// _PreLNLayer_51891794870651
// MI455X (gfx1250) — matrix operand order repaired and checked by reading, not compiled, not run on hardware
//
#include <hip/hip_runtime.h>

#define DEVINL __device__ __forceinline__

typedef __bf16 bf16_t;
typedef __bf16 v16bf __attribute__((ext_vector_type(16)));
typedef __bf16 v8bf  __attribute__((ext_vector_type(8)));
typedef __bf16 v4bf  __attribute__((ext_vector_type(4)));
typedef float  v8f   __attribute__((ext_vector_type(8)));
typedef unsigned int u32x4 __attribute__((ext_vector_type(4)));
typedef int i32x4 __attribute__((ext_vector_type(4)));
typedef int i32x8 __attribute__((ext_vector_type(8)));

constexpr int kB = 2, kS = 2048, kD = 1024, kH = 16, kHD = 64, kDFF = 4096;
constexpr int kM = kB * kS;           // 4096 token rows
constexpr float kEPS = 1e-5f;

// ---- WMMA fragment loaders (ISA 7.12.2 layouts, wave32) -------------------
// A (16x32 bf16): lane r=lane%16 holds row r; half=lane/16.
//   elems 0..7  -> k = kbase + half*8 + 0..7        (16B contiguous)
//   elems 8..15 -> k = kbase + 16 + half*8 + 0..7   (16B contiguous)
DEVINL v16bf load_frag_a(const bf16_t* base, int stride, int kbase) {
  int lane = threadIdx.x & 31;
  int r = lane & 15, half = lane >> 4;
  const bf16_t* p = base + (size_t)r * stride + kbase + half * 8;
  union { v16bf v; v8bf h[2]; } u;
  u.h[0] = *(const v8bf*)(p);
  u.h[1] = *(const v8bf*)(p + 16);
  return u.v;
}

// Same A-fragment pattern from an LDS tile with row stride 64 elems.
DEVINL v16bf load_frag_a_lds(const bf16_t* base, int kbase) {
  int lane = threadIdx.x & 31;
  int r = lane & 15, half = lane >> 4;
  const bf16_t* p = base + r * 64 + kbase + half * 8;
  union { v16bf v; v8bf h[2]; } u;
  u.h[0] = *(const v8bf*)(p);
  u.h[1] = *(const v8bf*)(p + 16);
  return u.v;
}

// B (32x16 bf16) fed from rows of an [N x K] matrix (B[k][n] = W[n][k]):
// lane n=lane%16 holds column n; elems 0..7 -> k = kbase + half*8 + 0..7, elems 8..15 -> k = kbase + 16 + half*8 + 0..7.
DEVINL v16bf load_frag_b(const bf16_t* base, int stride, int kbase) {
  int lane = threadIdx.x & 31;
  int n = lane & 15, half = lane >> 4;
  const bf16_t* p = base + (size_t)n * stride + kbase + half * 8;
  union { v16bf v; v8bf h[2]; } u;
  u.h[0] = *(const v8bf*)(p);
  u.h[1] = *(const v8bf*)(p + 16);
  return u.v;
}

// Same B-fragment pattern from an LDS tile with row stride 64 elems.
DEVINL v16bf load_frag_b_lds(const bf16_t* base, int kbase) {
  int lane = threadIdx.x & 31;
  int n = lane & 15, half = lane >> 4;
  const bf16_t* p = base + n * 64 + kbase + half * 8;
  union { v16bf v; v8bf h[2]; } u;
  u.h[0] = *(const v8bf*)(p);
  u.h[1] = *(const v8bf*)(p + 16);
  return u.v;
}

DEVINL v8f wmma_bf16(v16bf a, v16bf b, v8f c) {
  return __builtin_amdgcn_wmma_f32_16x16x32_bf16(false, a, false, b,
                                                 (short)0, c, false, false);
}

// ---- TDM: async 2-D tile load (global -> LDS), D# per ISA §8.3/8.4 --------
// Loads tile_h rows x tile_w bf16 elems (row stride = stride_elems) starting
// at gptr, packed row-major into LDS at lds_off. Tracked by TENSORcnt.
DEVINL void tdm_load_2d(const bf16_t* gptr, unsigned lds_off,
                        unsigned rowlen_elems, unsigned rows,
                        unsigned long long stride_elems,
                        unsigned tile_w, unsigned tile_h) {
  unsigned long long ga = (unsigned long long)gptr;
  u32x4 g0;
  g0[0] = 1u;                                        // count=1, user descriptor
  g0[1] = lds_off;                                   // lds_addr
  g0[2] = (unsigned)ga;                              // global_addr[31:0]
  g0[3] = (unsigned)((ga >> 32) & 0x1FFFFFFu) | (2u << 30);  // addr hi | type=2
  i32x8 g1;
  g1[0] = (int)(1u << 16);                           // data_size=1 (2 bytes)
  g1[1] = (int)((rowlen_elems & 0xFFFFu) << 16);     // tensor_dim0[15:0]
  g1[2] = (int)((rowlen_elems >> 16) | ((rows & 0xFFFFu) << 16)); // dim0 hi|dim1 lo
  g1[3] = (int)((rows >> 16) | (tile_w << 16));      // dim1 hi | tile_dim0
  g1[4] = (int)(tile_h & 0xFFFFu);                   // tile_dim1 (tile_dim2=0)
  g1[5] = (int)(unsigned)stride_elems;               // tensor_dim0_stride lo
  g1[6] = (int)((stride_elems >> 32) & 0xFFFFu);     // stride hi (dim1_stride=0)
  g1[7] = 0;
  i32x4 z4 = {0, 0, 0, 0};
  i32x8 z8 = {0, 0, 0, 0, 0, 0, 0, 0};
  __builtin_amdgcn_tensor_load_to_lds(g0, g1, z4, z4, z8, 0);
}

DEVINL unsigned lds_offset_of(const void* p) {
  return (unsigned)(unsigned long long)p;            // low 32 bits = LDS offset
}

// ---- fp32 -> bf16 convert --------------------------------------------------
__global__ void pl_cvt_bf16(const float* __restrict__ in, bf16_t* __restrict__ out, int n) {
  int i = blockIdx.x * blockDim.x + threadIdx.x;
  if (i < n) out[i] = (bf16_t)in[i];
}

// ---- LayerNorm (one 256-thread block per row of D=1024) -> bf16 ------------
__global__ void pl_layernorm(const float* __restrict__ x, const float* __restrict__ g,
                             const float* __restrict__ b, bf16_t* __restrict__ out) {
  int row = blockIdx.x;
  const float* xr = x + (size_t)row * kD;
  int t = threadIdx.x;
  float4 v = *(const float4*)(xr + t * 4);
  float s  = v.x + v.y + v.z + v.w;
  float s2 = v.x * v.x + v.y * v.y + v.z * v.z + v.w * v.w;
#pragma unroll
  for (int off = 1; off < 32; off <<= 1) {
    s  += __shfl_xor(s, off, 32);
    s2 += __shfl_xor(s2, off, 32);
  }
  __shared__ float rs[8], rs2[8];
  int wave = t >> 5, lane = t & 31;
  if (lane == 0) { rs[wave] = s; rs2[wave] = s2; }
  __syncthreads();
  float ts = 0.f, ts2 = 0.f;
#pragma unroll
  for (int i = 0; i < 8; ++i) { ts += rs[i]; ts2 += rs2[i]; }
  float mu = ts * (1.0f / kD);
  float var = ts2 * (1.0f / kD) - mu * mu;
  float rstd = rsqrtf(var + kEPS);
  float4 gg = *(const float4*)(g + t * 4);
  float4 bb = *(const float4*)(b + t * 4);
  v4bf o;
  o[0] = (bf16_t)((v.x - mu) * rstd * gg.x + bb.x);
  o[1] = (bf16_t)((v.y - mu) * rstd * gg.y + bb.y);
  o[2] = (bf16_t)((v.z - mu) * rstd * gg.z + bb.z);
  o[3] = (bf16_t)((v.w - mu) * rstd * gg.w + bb.w);
  *(v4bf*)(out + (size_t)row * kD + t * 4) = o;
}

// ---- WMMA GEMM: C[M,N] = A[M,K] @ W[N,K]^T + bias (+epilogue) --------------
// Block = 8 waves (4M x 2N), tile 128x128. Both A panel and W panel (128x64
// bf16 each) staged in LDS by TDM (wave0 loads A, wave1 loads W), double
// buffered. Fragments are read just-in-time from LDS to keep VGPR pressure
// ~90 (acc 64 + a 16 + b 8) and avoid scratch spills.
// mode 0: out bf16;  mode 1: out f32 = acc+bias+resid;  mode 2: out bf16 GELU
__global__ void pl_gemm(const bf16_t* __restrict__ A, const bf16_t* __restrict__ W,
                        const float* __restrict__ bias, const float* __restrict__ resid,
                        float* __restrict__ outF, bf16_t* __restrict__ outB,
                        int M, int N, int K, int mode) {
  __shared__ __align__(16) bf16_t asm_[2][128 * 64]; // A panels, 32 KB
  __shared__ __align__(16) bf16_t bsm[2][128 * 64];  // W panels, 32 KB
  int lane = threadIdx.x & 31;
  int half = lane >> 4, ln = lane & 15;
  int wave = threadIdx.x >> 5;
  int wm = wave >> 1, wn = wave & 1;                 // 4 (M) x 2 (N) waves
  int row0 = blockIdx.y * 128 + wm * 32;
  int rowB = blockIdx.y * 128;                       // block's first output row
  int colB = blockIdx.x * 128;                       // block's first output col
  int col0 = colB + wn * 64;

  const bf16_t* ap = A + (size_t)rowB * K;           // 128 activation rows
  const bf16_t* wp = W + (size_t)colB * K;           // 128 weight rows

  v8f acc[2][4] = {};
  int nt = K / 64;                                   // 64-wide K tiles

  // Prologue: async-load tile 0 of both panels.
  if (wave == 0) {
    tdm_load_2d(ap, lds_offset_of(&asm_[0][0]), K, 128, K, 64, 128);
  } else if (wave == 1) {
    tdm_load_2d(wp, lds_offset_of(&bsm[0][0]), K, 128, K, 64, 128);
  }
  if (wave < 2) __builtin_amdgcn_s_wait_tensorcnt(0);
  __syncthreads();

  for (int t = 0; t < nt; ++t) {
    int cur = t & 1;
    if (t + 1 < nt) {
      // Kick off async loads of the next K-tile; they overlap the 16 WMMAs.
      if (wave == 0) {
        tdm_load_2d(ap + (size_t)(t + 1) * 64, lds_offset_of(&asm_[1 - cur][0]),
                    K, 128, K, 64, 128);
      } else if (wave == 1) {
        tdm_load_2d(wp + (size_t)(t + 1) * 64, lds_offset_of(&bsm[1 - cur][0]),
                    K, 128, K, 64, 128);
      }
    }
#pragma unroll
    for (int kk = 0; kk < 64; kk += 32) {
      v16bf a0 = load_frag_a_lds(&asm_[cur][(wm * 32) * 64], kk);
      v16bf a1 = load_frag_a_lds(&asm_[cur][(wm * 32 + 16) * 64], kk);
#pragma unroll
      for (int tn = 0; tn < 4; ++tn) {
        v16bf bfr = load_frag_b_lds(&bsm[cur][(wn * 64 + tn * 16) * 64], kk);
        acc[0][tn] = wmma_bf16(a0, bfr, acc[0][tn]);
        acc[1][tn] = wmma_bf16(a1, bfr, acc[1][tn]);
      }
    }
    if (t + 1 < nt && wave < 2) __builtin_amdgcn_s_wait_tensorcnt(0);
    __syncthreads();
  }

#pragma unroll
  for (int tn = 0; tn < 4; ++tn) {
    int col = col0 + tn * 16 + ln;
    float bv = bias[col];
#pragma unroll
    for (int tm = 0; tm < 2; ++tm) {
#pragma unroll
      for (int i = 0; i < 8; ++i) {
        int row = row0 + tm * 16 + i + 8 * half;     // C layout: vgpr i, half
        size_t idx = (size_t)row * N + col;
        float v = acc[tm][tn][i] + bv;
        if (mode == 1) {
          outF[idx] = v + resid[idx];
        } else if (mode == 2) {
          outB[idx] = (bf16_t)(0.5f * v * (1.0f + erff(v * 0.70710678118f)));
        } else {
          outB[idx] = (bf16_t)v;
        }
      }
    }
  }
}

// ---- split heads: qkv[B,S,3D] -> Qh (scaled), Kh [B,H,S,HD], VhT [B,H,HD,S]-
__global__ void pl_split_heads(const bf16_t* __restrict__ qkv, bf16_t* __restrict__ Qh,
                               bf16_t* __restrict__ Kh, bf16_t* __restrict__ VhT) {
  int idx = blockIdx.x * blockDim.x + threadIdx.x;   // over B*S*D
  if (idx >= kB * kS * kD) return;
  int d = idx % kHD;
  int h = (idx / kHD) % kH;
  int s = (idx / kD) % kS;
  int b = idx / (kS * kD);
  const bf16_t* row = qkv + ((size_t)b * kS + s) * (3 * kD);
  float q = (float)row[h * kHD + d] * 0.125f;        // 1/sqrt(HD)
  bf16_t k = row[kD + h * kHD + d];
  bf16_t v = row[2 * kD + h * kHD + d];
  size_t bh = (size_t)b * kH + h;
  Qh[(bh * kS + s) * kHD + d] = (bf16_t)q;
  Kh[(bh * kS + s) * kHD + d] = k;
  VhT[(bh * kHD + d) * kS + s] = v;
}

// ---- Flash-style attention: one wave per (b,h, 16-query tile) --------------
__global__ void pl_attention(const bf16_t* __restrict__ Qh, const bf16_t* __restrict__ Kh,
                             const bf16_t* __restrict__ VhT, bf16_t* __restrict__ ctx) {
  __shared__ __align__(16) bf16_t plds[8][16 * 32];  // P tile per wave
  int wave = threadIdx.x >> 5;
  int lane = threadIdx.x & 31;
  int half = lane >> 4, ln = lane & 15;
  int gw = blockIdx.x * 8 + wave;
  int qt = gw % (kS / 16);
  int bh = gw / (kS / 16);

  const bf16_t* qbase = Qh + ((size_t)bh * kS + qt * 16) * kHD;
  v16bf qa0 = load_frag_a(qbase, kHD, 0);
  v16bf qa1 = load_frag_a(qbase, kHD, 32);

  v8f O[4] = {};
  float mrow[8], lrow[8];
#pragma unroll
  for (int i = 0; i < 8; ++i) { mrow[i] = -1e30f; lrow[i] = 0.f; }

  for (int kc = 0; kc < kS; kc += 32) {
    const bf16_t* kb = Kh + ((size_t)bh * kS + kc) * kHD;
    v8f s0 = {}, s1 = {};
    {
      v16bf b0 = load_frag_b(kb, kHD, 0);
      v16bf b1 = load_frag_b(kb, kHD, 32);
      s0 = wmma_bf16(qa0, b0, s0);
      s0 = wmma_bf16(qa1, b1, s0);
      v16bf b2 = load_frag_b(kb + (size_t)16 * kHD, kHD, 0);
      v16bf b3 = load_frag_b(kb + (size_t)16 * kHD, kHD, 32);
      s1 = wmma_bf16(qa0, b2, s1);
      s1 = wmma_bf16(qa1, b3, s1);
    }
#pragma unroll
    for (int i = 0; i < 8; ++i) {
      float rm = fmaxf(s0[i], s1[i]);
#pragma unroll
      for (int off = 1; off < 16; off <<= 1) rm = fmaxf(rm, __shfl_xor(rm, off, 32));
      float mn = fmaxf(mrow[i], rm);
      float sc = __expf(mrow[i] - mn);
      mrow[i] = mn;
      float p0 = __expf(s0[i] - mn);
      float p1 = __expf(s1[i] - mn);
      float rs = p0 + p1;
#pragma unroll
      for (int off = 1; off < 16; off <<= 1) rs += __shfl_xor(rs, off, 32);
      lrow[i] = lrow[i] * sc + rs;
      O[0][i] *= sc; O[1][i] *= sc; O[2][i] *= sc; O[3][i] *= sc;
      int r = i + 8 * half;                          // C-layout row of this vgpr
      plds[wave][r * 32 + ln]      = (bf16_t)p0;
      plds[wave][r * 32 + 16 + ln] = (bf16_t)p1;
    }
    // Re-load P in A-fragment layout (wave-local LDS, DS ops stay in order)
    union { v16bf v; v8bf h[2]; } up;
    up.h[0] = *(const v8bf*)&plds[wave][ln * 32 + half * 8];
    up.h[1] = *(const v8bf*)&plds[wave][ln * 32 + 16 + half * 8];
    v16bf pA = up.v;

    const bf16_t* vb = VhT + (size_t)bh * kHD * kS + kc;
#pragma unroll
    for (int t = 0; t < 4; ++t) {
      v16bf bv = load_frag_b(vb + (size_t)(t * 16) * kS, kS, 0);
      O[t] = wmma_bf16(pA, bv, O[t]);
    }
  }
  int b = bh / kH, h = bh % kH;
#pragma unroll
  for (int i = 0; i < 8; ++i) {
    int row = qt * 16 + i + 8 * half;
    float inv = 1.0f / lrow[i];
#pragma unroll
    for (int t = 0; t < 4; ++t) {
      int col = h * kHD + t * 16 + ln;
      ctx[((size_t)b * kS + row) * kD + col] = (bf16_t)(O[t][i] * inv);
    }
  }
}

// ---------------------------------------------------------------------------
extern "C" void kernel_launch(void* const* d_in, const int* in_sizes, int n_in,
                              void* d_out, int out_size, void* d_ws, size_t ws_size,
                              hipStream_t stream) {
  const float* x     = (const float*)d_in[0];
  const float* ln1_g = (const float*)d_in[1];
  const float* ln1_b = (const float*)d_in[2];
  const float* ln2_g = (const float*)d_in[3];
  const float* ln2_b = (const float*)d_in[4];
  const float* w_qkv = (const float*)d_in[5];
  const float* b_qkv = (const float*)d_in[6];
  const float* w_o   = (const float*)d_in[7];
  const float* b_o   = (const float*)d_in[8];
  const float* w1    = (const float*)d_in[9];
  const float* b1    = (const float*)d_in[10];
  const float* w2    = (const float*)d_in[11];
  const float* b2    = (const float*)d_in[12];

  char* cur = (char*)d_ws;
  auto alloc = [&](size_t bytes) -> void* {
    void* p = cur;
    cur += (bytes + 255) & ~(size_t)255;
    return p;
  };
  bf16_t* wqkv_b = (bf16_t*)alloc((size_t)3 * kD * kD * 2);
  bf16_t* wo_b   = (bf16_t*)alloc((size_t)kD * kD * 2);
  bf16_t* w1_b   = (bf16_t*)alloc((size_t)kDFF * kD * 2);
  bf16_t* w2_b   = (bf16_t*)alloc((size_t)kD * kDFF * 2);
  bf16_t* ln_buf = (bf16_t*)alloc((size_t)kM * kD * 2);       // ln1 then ln2
  bf16_t* qkv_b  = (bf16_t*)alloc((size_t)kM * 3 * kD * 2);
  bf16_t* Qh     = (bf16_t*)alloc((size_t)kB * kH * kS * kHD * 2);
  bf16_t* Kh     = (bf16_t*)alloc((size_t)kB * kH * kS * kHD * 2);
  bf16_t* VhT    = (bf16_t*)alloc((size_t)kB * kH * kHD * kS * 2);
  bf16_t* ctx    = (bf16_t*)alloc((size_t)kM * kD * 2);
  float*  x1     = (float*)alloc((size_t)kM * kD * 4);
  bf16_t* h_b    = qkv_b;  // FFN hidden (33.5MB) aliases dead qkv+Qh (25.2+8.4MB)

  // 1) weight conversions
  pl_cvt_bf16<<<(3 * kD * kD + 255) / 256, 256, 0, stream>>>(w_qkv, wqkv_b, 3 * kD * kD);
  pl_cvt_bf16<<<(kD * kD + 255) / 256, 256, 0, stream>>>(w_o, wo_b, kD * kD);
  pl_cvt_bf16<<<(kDFF * kD + 255) / 256, 256, 0, stream>>>(w1, w1_b, kDFF * kD);
  pl_cvt_bf16<<<(kD * kDFF + 255) / 256, 256, 0, stream>>>(w2, w2_b, kD * kDFF);

  // 2) LN1 -> bf16
  pl_layernorm<<<kM, 256, 0, stream>>>(x, ln1_g, ln1_b, ln_buf);

  // 3) QKV GEMM (M=4096, N=3072, K=1024) -> bf16
  dim3 g1(3 * kD / 128, kM / 128);
  pl_gemm<<<g1, 256, 0, stream>>>(ln_buf, wqkv_b, b_qkv, nullptr, nullptr, qkv_b,
                                  kM, 3 * kD, kD, 0);

  // 4) split heads (+ q scale, V transpose)
  pl_split_heads<<<(kB * kS * kD + 255) / 256, 256, 0, stream>>>(qkv_b, Qh, Kh, VhT);

  // 5) flash attention: B*H*(S/16)=4096 waves, 8 waves/block
  pl_attention<<<(kB * kH * (kS / 16)) / 8, 256, 0, stream>>>(Qh, Kh, VhT, ctx);

  // 6) O-proj + residual x -> x1 (f32)
  dim3 g2(kD / 128, kM / 128);
  pl_gemm<<<g2, 256, 0, stream>>>(ctx, wo_b, b_o, x, x1, nullptr, kM, kD, kD, 1);

  // 7) LN2 -> bf16
  pl_layernorm<<<kM, 256, 0, stream>>>(x1, ln2_g, ln2_b, ln_buf);

  // 8) FFN1 + GELU (M=4096, N=4096, K=1024) -> bf16
  dim3 g3(kDFF / 128, kM / 128);
  pl_gemm<<<g3, 256, 0, stream>>>(ln_buf, w1_b, b1, nullptr, nullptr, h_b,
                                  kM, kDFF, kD, 2);

  // 9) FFN2 + residual x1 -> d_out (f32) (M=4096, N=1024, K=4096)
  dim3 g4(kD / 128, kM / 128);
  pl_gemm<<<g4, 256, 0, stream>>>(h_b, w2_b, b2, x1, (float*)d_out, nullptr,
                                  kM, kD, kDFF, 1);

  (void)in_sizes; (void)n_in; (void)out_size; (void)ws_size;
}
